// Tokenflow_62466004353030
// MI455X (gfx1250) — compile-verified, matrix operand order checked by reading, not run on hardware
//
#include <hip/hip_runtime.h>
#include <hip/hip_bf16.h>
#include <stdint.h>

// Problem constants (from the reference)
#define V_ 256
#define T_ 256
#define K_ 36
#define L_ 32
#define D_ 1024

// Macro-tile geometry: one workgroup = 4 v's (144 rows) x 4 t's (128 cols).
#define BM 144
#define BN 128
#define KC 32                 // k-chunk per stage (one 16x16x32 WMMA k-step)
#define NSTAGE (D_ / KC)      // 32
#define LDSTRIDE 40           // ushorts per LDS tile row: 32 data + 8 pad = 80B (16B aligned, conflict-free)

typedef __attribute__((ext_vector_type(16))) __bf16 v16bf;
typedef __attribute__((ext_vector_type(8)))  float  v8f;
typedef __attribute__((ext_vector_type(4)))  unsigned int u32x4;
typedef __attribute__((ext_vector_type(8)))  int i32x8;
typedef __attribute__((ext_vector_type(4)))  int i32x4;

// ------------------------------------------------------------------
// Kernel 1: one-time fp32 -> bf16 downcast into workspace (HBM/L2 bound)
// ------------------------------------------------------------------
__global__ void cvt_f32_bf16(const float* __restrict__ src,
                             unsigned short* __restrict__ dst, int n4) {
  int i      = blockIdx.x * blockDim.x + threadIdx.x;
  int stride = gridDim.x * blockDim.x;
  for (; i < n4; i += stride) {
    float4 f = reinterpret_cast<const float4*>(src)[i];
    union { __bf16 h[4]; uint2 u; } p;
    p.h[0] = (__bf16)f.x; p.h[1] = (__bf16)f.y;
    p.h[2] = (__bf16)f.z; p.h[3] = (__bf16)f.w;
    reinterpret_cast<uint2*>(dst)[i] = p.u;
  }
}

// ------------------------------------------------------------------
// TDM: load a [rows x 32] bf16 tile (row stride 1024 elements in memory)
// into LDS with a 4-dword pad after every 16 dwords (row stride 80B in LDS).
// D# packing per CDNA5 ISA ch.8 (group0/group1; 2-D tensor, groups 2/3 zero).
// This toolchain's builtin is the 6-arg (clang-23 / therock-10.0) form.
// ------------------------------------------------------------------
__device__ __forceinline__ void tdm_load_tile(uint64_t gaddr, uint32_t lds_off,
                                              uint32_t tile_rows,
                                              uint32_t tensor_rows) {
  u32x4 g0;
  g0[0] = 1u;                                               // count=1, user mode, no gather
  g0[1] = lds_off;                                          // lds_addr (bytes)
  g0[2] = (uint32_t)gaddr;                                  // global_addr[31:0]
  g0[3] = ((uint32_t)(gaddr >> 32) & 0x1FFFFFFu) | (2u << 30); // global_addr[56:32] | type=2
  i32x8 g1;
  g1[0] = (1 << 16)      // data_size = 2 bytes
        | (1 << 20)      // pad_enable
        | (3 << 22)      // pad_interval: 16 dwords (=64B = one tile row)
        | (3 << 25);     // pad_amount:  4 dwords (row stride 80B in LDS)
  g1[1] = (int)(1024u << 16);          // tensor_dim0 = 1024 (bits 63:48)
  g1[2] = (int)(tensor_rows << 16);    // tensor_dim0 hi=0 ; tensor_dim1 lo16
  g1[3] = (int)(32u << 16);            // tensor_dim1 hi=0 ; tile_dim0 = 32
  g1[4] = (int)tile_rows;              // tile_dim1 ; tile_dim2 = 0
  g1[5] = 1024;                        // tensor_dim0_stride (elements)
  g1[6] = 0;
  g1[7] = 0;
  i32x4 gz4 = {0, 0, 0, 0};
  i32x8 gz8 = {0, 0, 0, 0, 0, 0, 0, 0};
  __builtin_amdgcn_tensor_load_to_lds(g0, g1, gz4, gz4, gz8, 0);
}

// ------------------------------------------------------------------
// Kernel 2: WMMA bf16 GEMM + fused Frobenius reduction.
// Grid (V/4, T/4), 192 threads = 6 waves (3 row-waves x 2 col-waves).
// ------------------------------------------------------------------
__global__ __launch_bounds__(192)
void simsq_kernel(const unsigned short* __restrict__ Aw,   // [9216][1024] bf16
                  const unsigned short* __restrict__ Bw,   // [8192][1024] bf16
                  float* __restrict__ out) {               // [V][T]
  __shared__ __align__(16) unsigned short Ash[2][BM * LDSTRIDE];
  __shared__ __align__(16) unsigned short Bsh[2][BN * LDSTRIDE];
  __shared__ float Osh[16];

  const int tid  = threadIdx.x;
  const int wave = tid >> 5;
  const int lane = tid & 31;
  const int n16  = lane & 15;
  const int hi   = lane >> 4;
  const int wr   = wave % 3;   // 48-row band within macro-tile
  const int wc   = wave / 3;   // 64-col band within macro-tile

  const int row0 = blockIdx.x * BM;   // global sim-row base (v*36+k space)
  const int col0 = blockIdx.y * BN;   // global sim-col base (t*32+l space)

  if (tid < 16) Osh[tid] = 0.0f;

  v8f acc[3][4];
  const v8f vzero = {};
#pragma unroll
  for (int i = 0; i < 3; ++i)
#pragma unroll
    for (int j = 0; j < 4; ++j) acc[i][j] = vzero;

  auto issue = [&](int s, int buf) {
    uint64_t ga = (uint64_t)(uintptr_t)Aw + ((uint64_t)row0 * D_ + (uint64_t)s * KC) * 2u;
    tdm_load_tile(ga, (uint32_t)(uintptr_t)(&Ash[buf][0]), BM, V_ * K_);
    uint64_t gb = (uint64_t)(uintptr_t)Bw + ((uint64_t)col0 * D_ + (uint64_t)s * KC) * 2u;
    tdm_load_tile(gb, (uint32_t)(uintptr_t)(&Bsh[buf][0]), BN, T_ * L_);
  };

  if (tid < 32) issue(0, 0);   // wave 0 drives the TDM (EXEC-independent, per-wave issue)

  for (int s = 0; s < NSTAGE; ++s) {
    if (tid < 32) __builtin_amdgcn_s_wait_tensorcnt(0);  // stage s DMA complete (wave 0)
    __syncthreads();                                     // ...visible to all; prev buf consumed
    if (tid < 32 && (s + 1) < NSTAGE) issue(s + 1, (s + 1) & 1);  // overlap DMA with compute

    const int buf = s & 1;

    // A fragments: lane holds row (lane&15); K chunks per bf16 16x32 layout:
    // lanes 0-15 -> K[0..7],[16..23]; lanes 16-31 -> K[8..15],[24..31].
    v16bf af[3];
#pragma unroll
    for (int ti = 0; ti < 3; ++ti) {
      int r = wr * 48 + ti * 16 + n16;
      const uint4* p = reinterpret_cast<const uint4*>(&Ash[buf][r * LDSTRIDE + hi * 8]);
      union { uint4 q[2]; v16bf v; } u;
      u.q[0] = p[0];      // +0B  : 8 bf16
      u.q[1] = p[2];      // +32B : 8 bf16
      af[ti] = u.v;
    }
    // B fragments: mirrored layout (lane holds col n = lane&15, K across VGPRs)
    v16bf bfr[4];
#pragma unroll
    for (int tj = 0; tj < 4; ++tj) {
      int r = wc * 64 + tj * 16 + n16;
      const uint4* p = reinterpret_cast<const uint4*>(&Bsh[buf][r * LDSTRIDE + hi * 8]);
      union { uint4 q[2]; v16bf v; } u;
      u.q[0] = p[0];
      u.q[1] = p[2];
      bfr[tj] = u.v;
    }

#pragma unroll
    for (int ti = 0; ti < 3; ++ti)
#pragma unroll
      for (int tj = 0; tj < 4; ++tj)
        acc[ti][tj] = __builtin_amdgcn_wmma_f32_16x16x32_bf16(
            false, af[ti], false, bfr[tj], (short)0, acc[ti][tj], false, false);
  }

  // Epilogue: sim^2 reduction. Wave's 48-row band spans exactly two v's
  // (boundary at block-local row 36*(wr+1)); each 16-col tile lies in one t.
  float p4[4] = {0.f, 0.f, 0.f, 0.f};
  const int boundary = 36 * (wr + 1);
#pragma unroll
  for (int ti = 0; ti < 3; ++ti)
#pragma unroll
    for (int tj = 0; tj < 4; ++tj) {
      const int tsel = tj >> 1;
#pragma unroll
      for (int j = 0; j < 8; ++j) {
        int m = wr * 48 + ti * 16 + hi * 8 + j;   // C element: M = j + 8*hi
        int vsel = (m < boundary) ? 0 : 1;
        float c = acc[ti][tj][j];
        p4[vsel * 2 + tsel] += c * c;
      }
    }
#pragma unroll
  for (int k = 0; k < 4; ++k) {
    float v = p4[k];
    for (int m = 16; m >= 1; m >>= 1) v += __shfl_xor(v, m, 32);
    p4[k] = v;
  }
  if (lane == 0) {
    const int vlo = wr, tA = wc * 2;
    atomicAdd(&Osh[(vlo + 0) * 4 + tA + 0], p4[0]);
    atomicAdd(&Osh[(vlo + 0) * 4 + tA + 1], p4[1]);
    atomicAdd(&Osh[(vlo + 1) * 4 + tA + 0], p4[2]);
    atomicAdd(&Osh[(vlo + 1) * 4 + tA + 1], p4[3]);
  }
  __syncthreads();
  if (tid < 16) {
    int vloc = tid >> 2, tloc = tid & 3;
    out[(blockIdx.x * 4 + vloc) * T_ + (blockIdx.y * 4 + tloc)] = Osh[tid];
  }
}

// ------------------------------------------------------------------
extern "C" void kernel_launch(void* const* d_in, const int* in_sizes, int n_in,
                              void* d_out, int out_size, void* d_ws, size_t ws_size,
                              hipStream_t stream) {
  const float* imgs = (const float*)d_in[0];   // [V,K,D] f32
  const float* caps = (const float*)d_in[1];   // [T,L,D] f32
  float* out = (float*)d_out;                  // [V,T] f32

  // Workspace: bf16 copies of A=[V*K,D] and B=[T*L,D] (35.6 MB total).
  unsigned short* Aw = (unsigned short*)d_ws;
  unsigned short* Bw = Aw + (size_t)V_ * K_ * D_;

  const int nA4 = V_ * K_ * D_ / 4;
  const int nB4 = T_ * L_ * D_ / 4;
  cvt_f32_bf16<<<1024, 256, 0, stream>>>(imgs, Aw, nA4);
  cvt_f32_bf16<<<1024, 256, 0, stream>>>(caps, Bw, nB4);

  dim3 grid(V_ / 4, T_ / 4);   // 64 x 64 macro-tiles, zero padding waste
  simsq_kernel<<<grid, 192, 0, stream>>>(Aw, Bw, out);
}
